// DynamicNeuralChild_3066606650036
// MI455X (gfx1250) — matrix operand order repaired and checked by reading, not compiled, not run on hardware
//
#include <hip/hip_runtime.h>

// ---------------------------------------------------------------------------
// Problem constants (from reference): S=1024, B=8, E=768, D=1024, H=8, DH=128,
// F=2048, L=4.  Tokens N = S*B = 8192.
// ---------------------------------------------------------------------------
#define TOK   8192
#define SEQ   1024
#define BATCH 8
#define EMB   768
#define DIM   1024
#define HEADS 8
#define DHEAD 128
#define FFN   2048
#define NLAY  4

typedef __attribute__((ext_vector_type(16))) __bf16 v16bf;
typedef __attribute__((ext_vector_type(8)))  float  v8f;
typedef int v4i __attribute__((vector_size(16)));   // matches builtin param type

union FragBF {
    uint4  q[2];   // 32 bytes = 16 bf16 = 8 VGPRs
    v16bf  v;
};

// ---------------------------------------------------------------------------
// CDNA5 async global->LDS copy (ASYNCcnt path). Guarded so the file still
// compiles if the toolchain lacks the builtin (falls back to batched
// load/store staging through VGPRs).
// ---------------------------------------------------------------------------
#define ASG __attribute__((address_space(1)))
#define ASL __attribute__((address_space(3)))

#if defined(__has_builtin)
#if __has_builtin(__builtin_amdgcn_global_load_async_to_lds_b128)
#define USE_ASYNC_LDS 1
#else
#define USE_ASYNC_LDS 0
#endif
#else
#define USE_ASYNC_LDS 0
#endif

#if USE_ASYNC_LDS
__device__ __forceinline__ void async_copy_b128(const __bf16* g, __bf16* l) {
    // Flat global address == AS1 address; flat LDS address low 32 bits == LDS
    // offset (ISA aperture layout), so inttoptr casts are the right bridge.
    __builtin_amdgcn_global_load_async_to_lds_b128(
        (ASG v4i*)(unsigned long long)(const __bf16*)g,
        (ASL v4i*)(unsigned long long)(__bf16*)l,
        0, 0);
}
__device__ __forceinline__ void wait_async0() {
#if __has_builtin(__builtin_amdgcn_s_wait_asynccnt)
    __builtin_amdgcn_s_wait_asynccnt(0);
#else
    asm volatile("s_wait_asynccnt 0x0" ::: "memory");
#endif
}
#endif

// ---------------------------------------------------------------------------
// fp32 -> bf16 elementwise conversion (grid-stride)
// ---------------------------------------------------------------------------
__global__ __launch_bounds__(256)
void cvt_f32_bf16_kernel(const float* __restrict__ in, __bf16* __restrict__ out, long n) {
    long i = (long)blockIdx.x * 256 + threadIdx.x;
    long stride = (long)gridDim.x * 256;
    for (; i < n; i += stride) out[i] = (__bf16)in[i];
}

// ---------------------------------------------------------------------------
// Tiled WMMA GEMM:  C[N,M] = act( A[N,K](bf16) * W[M,K]^T(bf16) + bias[M] )
// Block tile 128x128, BK=64. 256 threads = 8 waves in a 4x2 grid; each wave
// owns a 32x64 patch = 2x4 WMMA 16x16 tiles (64 accumulator VGPRs).
// Staging: async global->LDS (b128 per lane, ASYNCcnt) when available.
// ---------------------------------------------------------------------------
#define GBK  64
#define GLDT 72   // padded LDS row (64 + 8) elements -> 144B, 16B aligned

__global__ __launch_bounds__(256)
void gemm_bf16_wmma_kernel(const __bf16* __restrict__ A,
                           const __bf16* __restrict__ W,
                           const float*  __restrict__ bias,
                           float*        __restrict__ Cf,
                           __bf16*       __restrict__ Cb,
                           int N, int M, int K, int relu)
{
    __shared__ __align__(16) __bf16 Alds[128 * GLDT];
    __shared__ __align__(16) __bf16 Wlds[128 * GLDT];

    const int tid  = threadIdx.x;
    const int wave = tid >> 5;
    const int lane = tid & 31;
    const int ln   = lane & 15;
    const int hf   = lane >> 4;

    const int rowBlk = blockIdx.x * 128;
    const int colBlk = blockIdx.y * 128;
    const int wr = wave >> 1;      // 0..3 -> rows wr*32
    const int wc = wave & 1;       // 0..1 -> cols wc*64

    v8f acc[2][4] = {};

    // cooperative-load assignment: each thread moves 32 contiguous bf16 (4xb128)
    // for both the A tile and the W tile.
    const int lrow = tid >> 1;              // 0..127
    const int lcol = (tid & 1) * 32;        // 0 or 32
    const __bf16* aptr = A + (size_t)(rowBlk + lrow) * K + lcol;
    const __bf16* wptr = W + (size_t)(colBlk + lrow) * K + lcol;
    __bf16* alds = &Alds[lrow * GLDT + lcol];
    __bf16* wlds = &Wlds[lrow * GLDT + lcol];

    for (int k0 = 0; k0 < K; k0 += GBK) {
#if USE_ASYNC_LDS
#pragma unroll
        for (int qd = 0; qd < 4; ++qd)
            async_copy_b128(aptr + k0 + qd * 8, alds + qd * 8);
#pragma unroll
        for (int qd = 0; qd < 4; ++qd)
            async_copy_b128(wptr + k0 + qd * 8, wlds + qd * 8);
#else
        // batched fallback: all global loads issued before any LDS store so the
        // compiler can clause them and overlap latency.
        uint4 ta[4], tw[4];
#pragma unroll
        for (int qd = 0; qd < 4; ++qd)
            ta[qd] = *(const uint4*)(const void*)(aptr + k0 + qd * 8);
#pragma unroll
        for (int qd = 0; qd < 4; ++qd)
            tw[qd] = *(const uint4*)(const void*)(wptr + k0 + qd * 8);
#pragma unroll
        for (int qd = 0; qd < 4; ++qd)
            *(uint4*)(void*)(alds + qd * 8) = ta[qd];
#pragma unroll
        for (int qd = 0; qd < 4; ++qd)
            *(uint4*)(void*)(wlds + qd * 8) = tw[qd];
#endif
        // prefetch next k-tile into the cache hierarchy while we compute
        if (k0 + GBK < K) {
            __builtin_prefetch(aptr + k0 + GBK, 0, 1);
            __builtin_prefetch(wptr + k0 + GBK, 0, 1);
        }
#if USE_ASYNC_LDS
        wait_async0();
#endif
        __syncthreads();

#pragma unroll
        for (int ks = 0; ks < 2; ++ks) {
            FragBF af[2], bf[4];
#pragma unroll
            for (int mt = 0; mt < 2; ++mt) {
                const __bf16* base = &Alds[(wr * 32 + mt * 16 + ln) * GLDT + ks * 32];
                af[mt].q[0] = *(const uint4*)(const void*)(base + hf * 8);
                af[mt].q[1] = *(const uint4*)(const void*)(base + 16 + hf * 8);
            }
#pragma unroll
            for (int nt = 0; nt < 4; ++nt) {
                const __bf16* base = &Wlds[(wc * 64 + nt * 16 + ln) * GLDT + ks * 32];
                bf[nt].q[0] = *(const uint4*)(const void*)(base + hf * 8);
                bf[nt].q[1] = *(const uint4*)(const void*)(base + 16 + hf * 8);
            }
#pragma unroll
            for (int mt = 0; mt < 2; ++mt)
#pragma unroll
                for (int nt = 0; nt < 4; ++nt)
                    acc[mt][nt] = __builtin_amdgcn_wmma_f32_16x16x32_bf16(
                        false, af[mt].v, false, bf[nt].v, (short)0, acc[mt][nt], false, false);
        }
        __syncthreads();
    }

    // epilogue: C layout -> lane holds col n = ln, rows r + 8*hf (r = VGPR idx)
#pragma unroll
    for (int mt = 0; mt < 2; ++mt)
#pragma unroll
        for (int nt = 0; nt < 4; ++nt) {
            const int col = colBlk + wc * 64 + nt * 16 + ln;
            const float bv = bias ? bias[col] : 0.0f;
#pragma unroll
            for (int r = 0; r < 8; ++r) {
                const int row = rowBlk + wr * 32 + mt * 16 + r + 8 * hf;
                float v = acc[mt][nt][r] + bv;
                if (relu) v = fmaxf(v, 0.0f);
                if (Cf) Cf[(size_t)row * M + col] = v;
                if (Cb) Cb[(size_t)row * M + col] = (__bf16)v;
            }
        }
}

// ---------------------------------------------------------------------------
// LayerNorm (+optional residual add, +optional ReLU), fp32 in, fp32+bf16 out.
// One 256-thread block per row; row length C is a template param (512/1024).
// ---------------------------------------------------------------------------
template <int C>
__global__ __launch_bounds__(256)
void layernorm_kernel(const float* __restrict__ X,
                      const float* __restrict__ R,       // residual (nullable)
                      const float* __restrict__ gamma,
                      const float* __restrict__ beta,
                      float*       __restrict__ outF,    // nullable
                      __bf16*      __restrict__ outB,
                      int relu)
{
    constexpr int PER = C / 256;
    const int row = blockIdx.x;
    const int tid = threadIdx.x;
    const int wave = tid >> 5, lane = tid & 31;
    const float* x = X + (size_t)row * C;
    const float* r = R ? (R + (size_t)row * C) : nullptr;

    float vals[PER];
    float s = 0.0f;
#pragma unroll
    for (int i = 0; i < PER; ++i) {
        const int c = tid + i * 256;
        float v = x[c] + (r ? r[c] : 0.0f);
        vals[i] = v;
        s += v;
    }
    __shared__ float red[8];
#pragma unroll
    for (int m = 16; m; m >>= 1) s += __shfl_xor(s, m, 32);
    if (lane == 0) red[wave] = s;
    __syncthreads();
    float tot = 0.0f;
#pragma unroll
    for (int i = 0; i < 8; ++i) tot += red[i];
    const float mean = tot * (1.0f / (float)C);
    __syncthreads();

    float q = 0.0f;
#pragma unroll
    for (int i = 0; i < PER; ++i) { const float d = vals[i] - mean; q += d * d; }
#pragma unroll
    for (int m = 16; m; m >>= 1) q += __shfl_xor(q, m, 32);
    if (lane == 0) red[wave] = q;
    __syncthreads();
    float qt = 0.0f;
#pragma unroll
    for (int i = 0; i < 8; ++i) qt += red[i];
    const float rs = rsqrtf(qt * (1.0f / (float)C) + 1e-5f);

#pragma unroll
    for (int i = 0; i < PER; ++i) {
        const int c = tid + i * 256;
        float y = (vals[i] - mean) * rs * gamma[c] + beta[c];
        if (relu) y = fmaxf(y, 0.0f);
        if (outF) outF[(size_t)row * C + c] = y;
        outB[(size_t)row * C + c] = (__bf16)y;
    }
}

// ---------------------------------------------------------------------------
// Flash attention (online softmax), bf16 in/out, fp32 accumulate.
// Grid: (S/128, B*H). Block: 256 thr = 8 waves; wave w owns 16 query rows.
// KV processed in tiles of 64 keys staged in LDS (K async row-major copy,
// V transposed through VGPRs). QK^T and P*V via v_wmma_f32_16x16x32_bf16.
// ---------------------------------------------------------------------------
#define QKV_ROW 3072

__global__ __launch_bounds__(256)
void flash_attn_kernel(const __bf16* __restrict__ qkv,   // (S*B, 3D) bf16
                       __bf16*       __restrict__ O)     // (S*B, D)  bf16
{
    const int tid  = threadIdx.x;
    const int wave = tid >> 5;
    const int lane = tid & 31;
    const int ln   = lane & 15;
    const int hf   = lane >> 4;

    const int bh = blockIdx.y;
    const int bb = bh >> 3;        // batch
    const int hh = bh & 7;         // head
    const int q0 = blockIdx.x * 128 + wave * 16;   // first query of this wave

    __shared__ __align__(16) __bf16 Klds[64][136];      // keys x dh (pad)
    __shared__ __align__(16) __bf16 Vlds[128][72];      // dh x keys (transposed)
    __shared__ __align__(16) __bf16 Plds[8][16][72];    // per-wave P staging

    // --- load Q fragments for all 4 k-steps (DH=128 = 4*32) -----------------
    FragBF qf[4];
    {
        const __bf16* qrow = qkv + ((size_t)(q0 + ln) * BATCH + bb) * QKV_ROW + hh * DHEAD;
#pragma unroll
        for (int ks = 0; ks < 4; ++ks) {
            qf[ks].q[0] = *(const uint4*)(const void*)(qrow + ks * 32 + hf * 8);
            qf[ks].q[1] = *(const uint4*)(const void*)(qrow + ks * 32 + 16 + hf * 8);
        }
    }

    v8f o_acc[8] = {};             // 16 x 128 output accumulator (8 dh-tiles)
    float m_i[8], l_i[8];
#pragma unroll
    for (int r = 0; r < 8; ++r) { m_i[r] = -1e30f; l_i[r] = 0.0f; }
    const float scale = 0.0883883476483184f;   // 1/sqrt(128)

    for (int kv0 = 0; kv0 < SEQ; kv0 += 64) {
        // ---- cooperative load: K tile (async, row-major) + V (transposed) --
        {
            const int krow = tid >> 2;             // 0..63
            const int cg   = (tid & 3) * 32;       // dh column group
            const __bf16* kptr = qkv + ((size_t)(kv0 + krow) * BATCH + bb) * QKV_ROW
                                 + DIM + hh * DHEAD + cg;
#if USE_ASYNC_LDS
#pragma unroll
            for (int qd = 0; qd < 4; ++qd)
                async_copy_b128(kptr + qd * 8, &Klds[krow][cg + qd * 8]);
#else
            uint4 kt[4];
#pragma unroll
            for (int qd = 0; qd < 4; ++qd)
                kt[qd] = *(const uint4*)(const void*)(kptr + qd * 8);
#pragma unroll
            for (int qd = 0; qd < 4; ++qd)
                *(uint4*)(void*)&Klds[krow][cg + qd * 8] = kt[qd];
#endif
            const __bf16* vptr = qkv + ((size_t)(kv0 + krow) * BATCH + bb) * QKV_ROW
                                 + 2 * DIM + hh * DHEAD + cg;
            __bf16 vtmp[32];
            *(uint4*)(void*)&vtmp[0]  = *(const uint4*)(const void*)(vptr);
            *(uint4*)(void*)&vtmp[8]  = *(const uint4*)(const void*)(vptr + 8);
            *(uint4*)(void*)&vtmp[16] = *(const uint4*)(const void*)(vptr + 16);
            *(uint4*)(void*)&vtmp[24] = *(const uint4*)(const void*)(vptr + 24);
#pragma unroll
            for (int e = 0; e < 32; ++e) Vlds[cg + e][krow] = vtmp[e];
        }
#if USE_ASYNC_LDS
        wait_async0();
#endif
        __syncthreads();

        // ---- scores S = Q K^T (16 x 64) ------------------------------------
        v8f sc[4];
#pragma unroll
        for (int j = 0; j < 4; ++j) {
            v8f c = {};
#pragma unroll
            for (int ks = 0; ks < 4; ++ks) {
                FragBF kb;
                const __bf16* base = &Klds[j * 16 + ln][ks * 32];
                kb.q[0] = *(const uint4*)(const void*)(base + hf * 8);
                kb.q[1] = *(const uint4*)(const void*)(base + 16 + hf * 8);
                c = __builtin_amdgcn_wmma_f32_16x16x32_bf16(
                        false, qf[ks].v, false, kb.v, (short)0, c, false, false);
            }
            sc[j] = c;
        }
#pragma unroll
        for (int j = 0; j < 4; ++j)
#pragma unroll
            for (int r = 0; r < 8; ++r) sc[j][r] *= scale;

        // ---- online softmax (row reductions across the 16-lane half) -------
#pragma unroll
        for (int r = 0; r < 8; ++r) {
            float mx = fmaxf(fmaxf(sc[0][r], sc[1][r]), fmaxf(sc[2][r], sc[3][r]));
#pragma unroll
            for (int m2 = 1; m2 < 16; m2 <<= 1) mx = fmaxf(mx, __shfl_xor(mx, m2, 32));
            const float mnew = fmaxf(m_i[r], mx);
            const float corr = __expf(m_i[r] - mnew);
            l_i[r] *= corr;
#pragma unroll
            for (int j2 = 0; j2 < 8; ++j2) o_acc[j2][r] *= corr;
            float rsum = 0.0f;
#pragma unroll
            for (int j = 0; j < 4; ++j) {
                const float p = __expf(sc[j][r] - mnew);
                sc[j][r] = p;
                rsum += p;
            }
#pragma unroll
            for (int m2 = 1; m2 < 16; m2 <<= 1) rsum += __shfl_xor(rsum, m2, 32);
            l_i[r] += rsum;
            m_i[r] = mnew;
        }

        // ---- stage P into per-wave LDS (C layout -> memory row-major) ------
#pragma unroll
        for (int j = 0; j < 4; ++j)
#pragma unroll
            for (int r = 0; r < 8; ++r)
                Plds[wave][r + 8 * hf][j * 16 + ln] = (__bf16)sc[j][r];
        // same-wave DS ops are in-order: safe to read back immediately.

        // ---- O += P V  (contract over 64 keys = 2 k-steps) -----------------
#pragma unroll
        for (int kk = 0; kk < 2; ++kk) {
            FragBF pa;
            const __bf16* pbase = &Plds[wave][ln][kk * 32];
            pa.q[0] = *(const uint4*)(const void*)(pbase + hf * 8);
            pa.q[1] = *(const uint4*)(const void*)(pbase + 16 + hf * 8);
#pragma unroll
            for (int j2 = 0; j2 < 8; ++j2) {
                FragBF vb;
                const __bf16* vbase = &Vlds[j2 * 16 + ln][kk * 32];
                vb.q[0] = *(const uint4*)(const void*)(vbase + hf * 8);
                vb.q[1] = *(const uint4*)(const void*)(vbase + 16 + hf * 8);
                o_acc[j2] = __builtin_amdgcn_wmma_f32_16x16x32_bf16(
                                false, pa.v, false, vb.v, (short)0, o_acc[j2], false, false);
            }
        }
        __syncthreads();
    }

    // ---- normalize and write O (bf16) --------------------------------------
#pragma unroll
    for (int r = 0; r < 8; ++r) {
        const float inv = 1.0f / l_i[r];
        const int s = q0 + r + 8 * hf;
        __bf16* orow = O + ((size_t)s * BATCH + bb) * DIM + hh * DHEAD;
#pragma unroll
        for (int j2 = 0; j2 < 8; ++j2)
            orow[j2 * 16 + ln] = (__bf16)(o_acc[j2][r] * inv);
    }
}

// ---------------------------------------------------------------------------
// Host side: orchestrate the full network on `stream`.
// ---------------------------------------------------------------------------
extern "C" void kernel_launch(void* const* d_in, const int* in_sizes, int n_in,
                              void* d_out, int out_size, void* d_ws, size_t ws_size,
                              hipStream_t stream) {
    (void)in_sizes; (void)n_in; (void)out_size; (void)ws_size;

    const float* x        = (const float*)d_in[0];
    const float* enc_w    = (const float*)d_in[1];
    const float* enc_b    = (const float*)d_in[2];
    const float* enc_ln_g = (const float*)d_in[3];
    const float* enc_ln_b = (const float*)d_in[4];
    const float* qkv_w    = (const float*)d_in[5];
    const float* qkv_b    = (const float*)d_in[6];
    const float* out_w    = (const float*)d_in[7];
    const float* out_b    = (const float*)d_in[8];
    const float* ln1_g    = (const float*)d_in[9];
    const float* ln1_b    = (const float*)d_in[10];
    const float* ff1_w    = (const float*)d_in[11];
    const float* ff1_b    = (const float*)d_in[12];
    const float* ff2_w    = (const float*)d_in[13];
    const float* ff2_b    = (const float*)d_in[14];
    const float* ln2_g    = (const float*)d_in[15];
    const float* ln2_b    = (const float*)d_in[16];
    const float* dec_w1   = (const float*)d_in[17];
    const float* dec_b1   = (const float*)d_in[18];
    const float* dec_ln_g = (const float*)d_in[19];
    const float* dec_ln_b = (const float*)d_in[20];
    const float* dec_w2   = (const float*)d_in[21];
    const float* dec_b2   = (const float*)d_in[22];

    char* ws = (char*)d_ws;
    size_t off = 0;
    auto carve = [&](size_t bytes) -> void* {
        void* p = ws + off;
        off += (bytes + 255) & ~(size_t)255;
        return p;
    };

    __bf16* xb   = (__bf16*)carve((size_t)TOK * EMB * 2);        // x in bf16
    __bf16* wb   = (__bf16*)carve((size_t)3 * DIM * DIM * 2);    // weight scratch (max 3D*D)
    float*  tmpf = (float*) carve((size_t)TOK * DIM * 4);        // pre-LN fp32 scratch
    float*  hf32 = (float*) carve((size_t)TOK * DIM * 4);        // h fp32 (residual stream)
    __bf16* hb   = (__bf16*)carve((size_t)TOK * DIM * 2);        // h bf16 (GEMM input)
    __bf16* qkvb = (__bf16*)carve((size_t)TOK * 3 * DIM * 2);    // qkv bf16
    __bf16* ob   = (__bf16*)carve((size_t)TOK * DIM * 2);        // attention output bf16
    __bf16* fb   = (__bf16*)carve((size_t)TOK * FFN * 2);        // FF intermediate bf16
    __bf16* decb = (__bf16*)carve((size_t)TOK * (DIM / 2) * 2);  // decoder hidden bf16

    auto cvt = [&](const float* src, __bf16* dst, long n) {
        int blocks = (int)((n + 255) / 256);
        if (blocks > 2048) blocks = 2048;
        cvt_f32_bf16_kernel<<<blocks, 256, 0, stream>>>(src, dst, n);
    };
    auto gemm = [&](const __bf16* A, const __bf16* W, const float* bias,
                    float* Cf, __bf16* Cb, int N, int M, int K, int relu) {
        dim3 grid(N / 128, M / 128);
        gemm_bf16_wmma_kernel<<<grid, 256, 0, stream>>>(A, W, bias, Cf, Cb, N, M, K, relu);
    };

    // ---------------- encoder MLP: relu(LN(x @ enc_w^T + b)) ----------------
    cvt(x, xb, (long)TOK * EMB);
    cvt(enc_w, wb, (long)DIM * EMB);
    gemm(xb, wb, enc_b, tmpf, nullptr, TOK, DIM, EMB, 0);
    layernorm_kernel<DIM><<<TOK, 256, 0, stream>>>(tmpf, nullptr, enc_ln_g, enc_ln_b,
                                                   hf32, hb, 1);

    // ---------------- 4 transformer encoder layers (post-norm) --------------
    for (int l = 0; l < NLAY; ++l) {
        // qkv projection
        cvt(qkv_w + (size_t)l * 3 * DIM * DIM, wb, (long)3 * DIM * DIM);
        gemm(hb, wb, qkv_b + (size_t)l * 3 * DIM, nullptr, qkvb, TOK, 3 * DIM, DIM, 0);

        // attention (flash, WMMA QK^T and PV)
        flash_attn_kernel<<<dim3(SEQ / 128, BATCH * HEADS), 256, 0, stream>>>(qkvb, ob);

        // output projection + residual + LN1
        cvt(out_w + (size_t)l * DIM * DIM, wb, (long)DIM * DIM);
        gemm(ob, wb, out_b + (size_t)l * DIM, tmpf, nullptr, TOK, DIM, DIM, 0);
        layernorm_kernel<DIM><<<TOK, 256, 0, stream>>>(tmpf, hf32,
                                                       ln1_g + (size_t)l * DIM,
                                                       ln1_b + (size_t)l * DIM,
                                                       hf32, hb, 0);

        // feed-forward
        cvt(ff1_w + (size_t)l * FFN * DIM, wb, (long)FFN * DIM);
        gemm(hb, wb, ff1_b + (size_t)l * FFN, nullptr, fb, TOK, FFN, DIM, 1);
        cvt(ff2_w + (size_t)l * DIM * FFN, wb, (long)DIM * FFN);
        gemm(fb, wb, ff2_b + (size_t)l * DIM, tmpf, nullptr, TOK, DIM, FFN, 0);
        layernorm_kernel<DIM><<<TOK, 256, 0, stream>>>(tmpf, hf32,
                                                       ln2_g + (size_t)l * DIM,
                                                       ln2_b + (size_t)l * DIM,
                                                       hf32, hb, 0);
    }

    // ---------------- decoder MLP -------------------------------------------
    cvt(dec_w1, wb, (long)(DIM / 2) * DIM);
    gemm(hb, wb, dec_b1, tmpf, nullptr, TOK, DIM / 2, DIM, 0);
    layernorm_kernel<DIM / 2><<<TOK, 256, 0, stream>>>(tmpf, nullptr, dec_ln_g, dec_ln_b,
                                                       nullptr, decb, 1);
    cvt(dec_w2, wb, (long)EMB * (DIM / 2));
    gemm(decb, wb, dec_b2, (float*)d_out, nullptr, TOK, EMB, DIM / 2, 0);
}
